// AttentiveFP_36249523978319
// MI455X (gfx1250) — compile-verified, matrix operand order checked by reading, not run on hardware
//
#include <hip/hip_runtime.h>
#include <hip/hip_bf16.h>
#include <math.h>

// ---- problem constants (match reference setup_inputs) ----
#define Dv   300      // feature dim
#define DP   320      // padded feature dim (multiple of 32 for WMMA K)
#define D3P  960      // padded 3*D (GRU gates)
#define NNn  50000    // nodes  (3125 * 16)
#define EEe  200000   // edges
#define NEc  250000   // edges + self loops (15625 * 16)
#define GGg  2000     // graphs (125 * 16)

typedef __attribute__((ext_vector_type(16))) _Float16 v16h;
typedef __attribute__((ext_vector_type(8)))  float    v8f;

union Frag16 { v16h v; float4 f4[2]; };

// Per-lane WMMA fragment load (16-bit A/B operand, 16x32 / 32x16 tile).
// lane l (0..31): row = base row + (l&15).
// halves 0..7  <-> K = k0 + (l>=16 ? 8 : 0) + 0..7
// halves 8..15 <-> K = k0 + 16 + (l>=16 ? 8 : 0) + 0..7
__device__ __forceinline__ v16h load_frag16(const _Float16* base, int ld,
                                            int row, int k0, int lane) {
  Frag16 u;
  const _Float16* p = base + (size_t)row * ld + k0 + ((lane >> 4) << 3);
  u.f4[0] = *reinterpret_cast<const float4*>(p);
  u.f4[1] = *reinterpret_cast<const float4*>(p + 16);
  return u.v;
}

// C = act(A[M,K] @ W[Nout,K]^T + bias), fp32 out, row stride Nout.
// One wave computes a 16x64 tile (4 accumulators, A fragment reused 4x).
// Requires Nout % 64 == 0 (all call sites use Nout==320). act: 0=none, 1=leaky(0.01)
__global__ void k_gemm_wmma(const _Float16* __restrict__ A,
                            const _Float16* __restrict__ W,
                            float* __restrict__ C,
                            int M, int K, int Nout,
                            const float* __restrict__ bias, int act) {
  int wid  = threadIdx.x >> 5;
  int lane = threadIdx.x & 31;
  int nT4  = Nout >> 6;                       // groups of 4 N-tiles
  int gw   = blockIdx.x * (blockDim.x >> 5) + wid;
  int total = (M >> 4) * nT4;
  if (gw >= total) return;
  int tM = gw / nT4;
  int t4 = gw - tM * nT4;

  v8f acc0 = {}, acc1 = {}, acc2 = {}, acc3 = {};
  int arow = (tM << 4) + (lane & 15);
  int wrow = (t4 << 6) + (lane & 15);
  for (int k0 = 0; k0 < K; k0 += 32) {
    v16h a  = load_frag16(A, K, arow, k0, lane);
    v16h b0 = load_frag16(W, K, wrow,      k0, lane);
    v16h b1 = load_frag16(W, K, wrow + 16, k0, lane);
    v16h b2 = load_frag16(W, K, wrow + 32, k0, lane);
    v16h b3 = load_frag16(W, K, wrow + 48, k0, lane);
    acc0 = __builtin_amdgcn_wmma_f32_16x16x32_f16(false, a, false, b0, (short)0, acc0, false, false);
    acc1 = __builtin_amdgcn_wmma_f32_16x16x32_f16(false, a, false, b1, (short)0, acc1, false, false);
    acc2 = __builtin_amdgcn_wmma_f32_16x16x32_f16(false, a, false, b2, (short)0, acc2, false, false);
    acc3 = __builtin_amdgcn_wmma_f32_16x16x32_f16(false, a, false, b3, (short)0, acc3, false, false);
  }
  int rbase = (tM << 4) + ((lane >> 4) << 3);
  v8f* accs[4] = { &acc0, &acc1, &acc2, &acc3 };
  for (int u = 0; u < 4; ++u) {
    int col = (t4 << 6) + (u << 4) + (lane & 15);
    float b0 = bias ? bias[col] : 0.f;
    for (int r = 0; r < 8; ++r) {
      float v = (*accs[u])[r] + b0;
      if (act == 1) v = v > 0.f ? v : 0.01f * v;
      C[(size_t)(rbase + r) * Nout + col] = v;
    }
  }
}

// Fused GRU: out = relu(GRU(elu_h16 @ Wih^T + bih, h16 @ Whh^T + bhh, h32)).
// Each wave owns a 16x16 column block and computes ALL THREE gates for both
// GEMMs (6 WMMA accumulator chains, A fragments reused 3x each), then the
// gate nonlinearity in registers. Writes fp32 result AND its f16 copy.
// Wih/Whh packed [960,320] f16, gate g rows at g*320. bih/bhh raw fp32 [900].
__global__ void k_gru_fused(const _Float16* __restrict__ Ain,   // elu(h) f16 [M,320]
                            const _Float16* __restrict__ Hin16, // state f16 [M,320]
                            const float*    __restrict__ Hin32, // state fp32 [M,320]
                            const _Float16* __restrict__ Wih,
                            const _Float16* __restrict__ Whh,
                            const float* __restrict__ bih, const float* __restrict__ bhh,
                            float* __restrict__ out32, _Float16* __restrict__ out16,
                            int M) {
  int wid  = threadIdx.x >> 5;
  int lane = threadIdx.x & 31;
  int gw   = blockIdx.x * (blockDim.x >> 5) + wid;
  int nTN  = DP >> 4;                          // 20 column tiles
  int total = (M >> 4) * nTN;
  if (gw >= total) return;
  int tM = gw / nTN;
  int tN = gw - tM * nTN;

  v8f ir_ = {}, iz_ = {}, in_ = {}, hr_ = {}, hz_ = {}, hn_ = {};
  int arow = (tM << 4) + (lane & 15);
  int wrow = (tN << 4) + (lane & 15);
  for (int k0 = 0; k0 < DP; k0 += 32) {
    v16h a1 = load_frag16(Ain,   DP, arow, k0, lane);
    v16h a2 = load_frag16(Hin16, DP, arow, k0, lane);
    v16h br = load_frag16(Wih, DP, wrow,          k0, lane);
    v16h bz = load_frag16(Wih, DP, wrow + DP,     k0, lane);
    v16h bn = load_frag16(Wih, DP, wrow + 2 * DP, k0, lane);
    ir_ = __builtin_amdgcn_wmma_f32_16x16x32_f16(false, a1, false, br, (short)0, ir_, false, false);
    iz_ = __builtin_amdgcn_wmma_f32_16x16x32_f16(false, a1, false, bz, (short)0, iz_, false, false);
    in_ = __builtin_amdgcn_wmma_f32_16x16x32_f16(false, a1, false, bn, (short)0, in_, false, false);
    v16h cr = load_frag16(Whh, DP, wrow,          k0, lane);
    v16h cz = load_frag16(Whh, DP, wrow + DP,     k0, lane);
    v16h cn = load_frag16(Whh, DP, wrow + 2 * DP, k0, lane);
    hr_ = __builtin_amdgcn_wmma_f32_16x16x32_f16(false, a2, false, cr, (short)0, hr_, false, false);
    hz_ = __builtin_amdgcn_wmma_f32_16x16x32_f16(false, a2, false, cz, (short)0, hz_, false, false);
    hn_ = __builtin_amdgcn_wmma_f32_16x16x32_f16(false, a2, false, cn, (short)0, hn_, false, false);
  }
  int col   = (tN << 4) + (lane & 15);
  int rbase = (tM << 4) + ((lane >> 4) << 3);
  if (col < Dv) {
    float b_ir = bih[col], b_iz = bih[Dv + col], b_in = bih[2 * Dv + col];
    float b_hr = bhh[col], b_hz = bhh[Dv + col], b_hn = bhh[2 * Dv + col];
    for (int r = 0; r < 8; ++r) {
      size_t idx = (size_t)(rbase + r) * DP + col;
      float rg = 1.f / (1.f + expf(-(ir_[r] + b_ir + hr_[r] + b_hr)));
      float zg = 1.f / (1.f + expf(-(iz_[r] + b_iz + hz_[r] + b_hz)));
      float ng = tanhf(in_[r] + b_in + rg * (hn_[r] + b_hn));
      float v  = (1.f - zg) * ng + zg * Hin32[idx];
      v = fmaxf(v, 0.f);
      out32[idx] = v;
      out16[idx] = (_Float16)v;
    }
  } else {
    for (int r = 0; r < 8; ++r) {
      size_t idx = (size_t)(rbase + r) * DP + col;
      out32[idx] = 0.f;
      out16[idx] = (_Float16)0.f;
    }
  }
}

// ---------------- helpers ----------------
__device__ __forceinline__ void atomicMaxFloat(float* addr, float val) {
  if (val >= 0.f) atomicMax((int*)addr, __float_as_int(val));
  else            atomicMin((unsigned int*)addr, __float_as_uint(val));
}

__global__ void k_fill(float* p, float v, long n) {
  long i = blockIdx.x * (long)blockDim.x + threadIdx.x;
  if (i < n) p[i] = v;
}

__global__ void k_pad_vec(float* dst, const float* src, int nvalid, int ntot) {
  int i = blockIdx.x * blockDim.x + threadIdx.x;
  if (i < ntot) dst[i] = (i < nvalid) ? src[i] : 0.f;
}

// Pack fp32 weight [chunks*300, K=300 at col0, row stride src_ld] -> f16 [chunks*320,320]
__global__ void k_pack_w(_Float16* dst, const float* src,
                         int src_ld, int col0, int chunks) {
  long i = blockIdx.x * (long)blockDim.x + threadIdx.x;
  long tot = (long)chunks * DP * DP;
  if (i >= tot) return;
  int  k  = (int)(i % DP);
  long r  = i / DP;
  int  c  = (int)(r / DP);
  int  rr = (int)(r % DP);
  float v = 0.f;
  if (rr < Dv && k < Dv)
    v = src[(size_t)(c * Dv + rr) * src_ld + col0 + k];
  dst[i] = (_Float16)v;
}

// xf0 = emb_atom[x0] + emb_chir[x1]  ->  f16 [N,320]
__global__ void k_embed(const int* __restrict__ x, const float* __restrict__ ea,
                        const float* __restrict__ ec, _Float16* __restrict__ dst) {
  long i = blockIdx.x * (long)blockDim.x + threadIdx.x;
  if (i >= (long)NNn * DP) return;
  int  j = (int)(i % DP);
  long n = i / DP;
  float v = 0.f;
  if (j < Dv) {
    int a = x[2 * n], c = x[2 * n + 1];
    v = ea[a * Dv + j] + ec[c * Dv + j];
  }
  dst[i] = (_Float16)v;
}

// fp32 [M,320] -> f16 [M,320]; optional bias (len>=300) and ELU (act==2)
__global__ void k_to_f16(const float* __restrict__ src,
                         const float* __restrict__ bias,
                         _Float16* __restrict__ dst, long M, int act) {
  long i = blockIdx.x * (long)blockDim.x + threadIdx.x;
  if (i >= M * DP) return;
  int j = (int)(i % DP);
  float v = 0.f;
  if (j < Dv) {
    v = src[i];
    if (bias) v += bias[j];
    if (act == 2) v = v > 0.f ? v : expf(v) - 1.f;
  }
  dst[i] = (_Float16)v;
}

// out[n] = dot(X[n, 0:300], vec)
__global__ void k_rowdot(const float* __restrict__ X, const float* __restrict__ vec,
                         float* __restrict__ out, int M) {
  int n = blockIdx.x * blockDim.x + threadIdx.x;
  if (n >= M) return;
  const float* r = X + (size_t)n * DP;
  float a = 0.f;
  for (int j = 0; j < Dv; ++j) a += r[j] * vec[j];
  out[n] = a;
}

// T[c, :] = (ge_e1[bt] + ge_e2[bd]) @ W1b^T, W1b = ge_lin1_W[:, 300:600]
__global__ void k_table_T(const float* __restrict__ ge1, const float* __restrict__ ge2,
                          const float* __restrict__ geW1, float* __restrict__ T) {
  int i = blockIdx.x * blockDim.x + threadIdx.x;
  if (i >= 15 * DP) return;
  int c = i / DP, j = i % DP;
  float v = 0.f;
  if (j < Dv) {
    int bt = c / 3, bd = c % 3;
    const float* wr = geW1 + (size_t)j * (2 * Dv) + Dv;
    for (int k = 0; k < Dv; ++k)
      v += wr[k] * (ge1[bt * Dv + k] + ge2[bd * Dv + k]);
  }
  T[i] = v;
}

// GATEConv edge pass: xj = leaky(P[src] + T[combo]); a = leaky(xj.attl + ar[dst])
__global__ void k_edge0_alpha(const int* __restrict__ ei, const int* __restrict__ ea,
                              const float* __restrict__ P, const float* __restrict__ T,
                              const float* __restrict__ ar, const float* __restrict__ attl,
                              _Float16* __restrict__ XJ, float* __restrict__ aE,
                              int* __restrict__ dst0, float* __restrict__ m) {
  int e = blockIdx.x * blockDim.x + threadIdx.x;
  if (e >= NEc) return;
  int s, d, bt, bd;
  if (e < EEe) { s = ei[e]; d = ei[EEe + e]; bt = ea[2 * e]; bd = ea[2 * e + 1]; }
  else         { s = d = e - EEe; bt = 4; bd = 0; }
  const float* Pp = P + (size_t)s * DP;
  const float* Tp = T + (size_t)(bt * 3 + bd) * DP;
  _Float16* xp = XJ + (size_t)e * DP;
  float acc = 0.f;
  for (int j = 0; j < Dv; ++j) {
    float v = Pp[j] + Tp[j];
    v = v > 0.f ? v : 0.01f * v;
    xp[j] = (_Float16)v;
    acc += v * attl[j];
  }
  for (int j = Dv; j < DP; ++j) xp[j] = (_Float16)0.f;
  float a = acc + ar[d];
  a = a > 0.f ? a : 0.01f * a;
  aE[e] = a;
  dst0[e] = d;
  atomicMaxFloat(&m[d], a);
}

// generic: a = leaky(avs[src?src[e]:e] + avd[dst[e]], slope); atomic max into m
__global__ void k_edge_max(const int* __restrict__ src, const int* __restrict__ dst,
                           const float* __restrict__ avs, const float* __restrict__ avd,
                           float* __restrict__ aE, float* __restrict__ m,
                           int E, float slope) {
  int e = blockIdx.x * blockDim.x + threadIdx.x;
  if (e >= E) return;
  int s = src ? src[e] : e;
  int d = dst[e];
  float a = avs[s] + avd[d];
  a = a > 0.f ? a : slope * a;
  aE[e] = a;
  atomicMaxFloat(&m[d], a);
}

__global__ void k_edge_exp(const float* __restrict__ aE, const int* __restrict__ dst,
                           const float* __restrict__ m, float* __restrict__ ssum,
                           float* __restrict__ eE, int E) {
  int e = blockIdx.x * blockDim.x + threadIdx.x;
  if (e >= E) return;
  int d = dst[e];
  float ex = expf(aE[e] - m[d]);
  eE[e] = ex;
  atomicAdd(&ssum[d], ex);
}

// S[dst[e]] += (eE/sum) * X[src?src[e]:e]   (fp32 source rows)
__global__ void k_accum_gather(const int* __restrict__ src, const int* __restrict__ dst,
                               const float* __restrict__ eE, const float* __restrict__ ssum,
                               const float* __restrict__ X, float* __restrict__ S, int E) {
  long t = blockIdx.x * (long)blockDim.x + threadIdx.x;
  long e = t / DP; int j = (int)(t % DP);
  if (e >= E || j >= Dv) return;
  int d = dst[e];
  float w = eE[e] / (ssum[d] + 1e-16f);
  int sv = src ? src[e] : (int)e;
  atomicAdd(&S[(size_t)d * DP + j], w * X[(size_t)sv * DP + j]);
}

// S[dst[e]] += (eE/sum) * X16[e]   (per-edge f16 rows, layer-0 xj)
__global__ void k_accum_direct16(const int* __restrict__ dst,
                                 const float* __restrict__ eE, const float* __restrict__ ssum,
                                 const _Float16* __restrict__ X16, float* __restrict__ S, int E) {
  long t = blockIdx.x * (long)blockDim.x + threadIdx.x;
  long e = t / DP; int j = (int)(t % DP);
  if (e >= E || j >= Dv) return;
  int d = dst[e];
  float w = eE[e] / (ssum[d] + 1e-16f);
  atomicAdd(&S[(size_t)d * DP + j], w * (float)X16[(size_t)e * DP + j]);
}

// readout init: O[batch[n]] += XF[n]
__global__ void k_accum_nodes(const int* __restrict__ batch, const float* __restrict__ XF,
                              float* __restrict__ O) {
  long t = blockIdx.x * (long)blockDim.x + threadIdx.x;
  if (t >= (long)NNn * DP) return;
  long n = t / DP; int j = (int)(t % DP);
  if (j >= Dv) return;
  atomicAdd(&O[(size_t)batch[n] * DP + j], XF[t]);
}

__global__ void k_relu_pad(float* p, long M) {
  long i = blockIdx.x * (long)blockDim.x + threadIdx.x;
  if (i >= M * DP) return;
  int j = (int)(i % DP);
  p[i] = (j < Dv) ? fmaxf(p[i], 0.f) : 0.f;
}

__global__ void k_writeout(const float* __restrict__ O, float* __restrict__ out) {
  long i = blockIdx.x * (long)blockDim.x + threadIdx.x;
  if (i >= (long)GGg * DP) return;
  long g = i / DP; int j = (int)(i % DP);
  if (j < Dv) out[g * Dv + j] = O[i];
}

__global__ void k_pred(const float* __restrict__ O, const float* __restrict__ w,
                       const float* __restrict__ b, float* __restrict__ out) {
  int g = blockIdx.x * blockDim.x + threadIdx.x;
  if (g >= GGg) return;
  const float* r = O + (size_t)g * DP;
  float a = 0.f;
  for (int j = 0; j < Dv; ++j) a += r[j] * w[j];
  out[(long)GGg * Dv + g] = a + b[0];
}

// --------------------------------------------------------------------------
static inline int blks(long n) { return (int)((n + 255) / 256); }

extern "C" void kernel_launch(void* const* d_in, const int* in_sizes, int n_in,
                              void* d_out, int out_size, void* d_ws, size_t ws_size,
                              hipStream_t stream) {
  // ---- inputs (setup_inputs order) ----
  const int*   x          = (const int*)  d_in[0];
  const int*   edge_index = (const int*)  d_in[1];
  const int*   edge_attr  = (const int*)  d_in[2];
  const int*   batch      = (const int*)  d_in[3];
  const float* emb_atom   = (const float*)d_in[4];
  const float* emb_chir   = (const float*)d_in[5];
  const float* lin1_W     = (const float*)d_in[6];
  const float* lin1_b     = (const float*)d_in[7];
  const float* ge_e1      = (const float*)d_in[8];
  const float* ge_e2      = (const float*)d_in[9];
  const float* ge_att_l   = (const float*)d_in[10];
  const float* ge_att_r   = (const float*)d_in[11];
  const float* ge_lin1_W  = (const float*)d_in[12];
  const float* ge_lin2_W  = (const float*)d_in[13];
  const float* ge_bias    = (const float*)d_in[14];
  const float* gat_W      = (const float*)d_in[15];
  const float* gat_att_s  = (const float*)d_in[16];
  const float* gat_att_d  = (const float*)d_in[17];
  const float* gat_b      = (const float*)d_in[18];
  const float* gru_wih    = (const float*)d_in[19];
  const float* gru_whh    = (const float*)d_in[20];
  const float* gru_bih    = (const float*)d_in[21];
  const float* gru_bhh    = (const float*)d_in[22];
  const float* mol_Ws     = (const float*)d_in[23];
  const float* mol_Wd     = (const float*)d_in[24];
  const float* mol_att_s  = (const float*)d_in[25];
  const float* mol_att_d  = (const float*)d_in[26];
  const float* mol_b      = (const float*)d_in[27];
  const float* mgru_wih   = (const float*)d_in[28];
  const float* mgru_whh   = (const float*)d_in[29];
  const float* mgru_bih   = (const float*)d_in[30];
  const float* mgru_bhh   = (const float*)d_in[31];
  const float* lin2_W     = (const float*)d_in[32];
  const float* lin2_b     = (const float*)d_in[33];
  float* out = (float*)d_out;

  // ---- workspace bump allocator ----
  char* base = (char*)d_ws;
  size_t off = 0;
  auto alloc = [&](size_t bytes) -> void* {
    void* p = base + off;
    off += (bytes + 255) & ~(size_t)255;
    return p;
  };
  float*     XF32a = (float*)    alloc((size_t)NNn * DP * 4);   // node state ping
  float*     XF32b = (float*)    alloc((size_t)NNn * DP * 4);   // node state pong
  _Float16*  XF16a = (_Float16*) alloc((size_t)NNn * DP * 2);
  _Float16*  XF16b = (_Float16*) alloc((size_t)NNn * DP * 2);
  float*     P32   = (float*)    alloc((size_t)NNn * DP * 4);
  float*     S32   = (float*)    alloc((size_t)NNn * DP * 4);
  _Float16*  TMP16 = (_Float16*) alloc((size_t)NNn * DP * 2);
  _Float16*  XJ    = (_Float16*) alloc((size_t)NEc * DP * 2);   // layer-0 edge features
  float*     H0    = (float*)    alloc((size_t)NNn * DP * 4);
  float*     O32a  = (float*)    alloc((size_t)GGg * DP * 4);
  float*     O32b  = (float*)    alloc((size_t)GGg * DP * 4);
  _Float16*  O16a  = (_Float16*) alloc((size_t)GGg * DP * 2);
  _Float16*  O16b  = (_Float16*) alloc((size_t)GGg * DP * 2);
  float*     XD    = (float*)    alloc((size_t)GGg * DP * 4);
  _Float16*  HG16  = (_Float16*) alloc((size_t)GGg * DP * 2);
  float*     aE    = (float*)    alloc((size_t)NEc * 4);
  float*     eEx   = (float*)    alloc((size_t)NEc * 4);
  int*       dst0  = (int*)      alloc((size_t)NEc * 4);
  float*     mseg  = (float*)    alloc((size_t)NNn * 4);
  float*     sseg  = (float*)    alloc((size_t)NNn * 4);
  float*     asN   = (float*)    alloc((size_t)NNn * 4);
  float*     adN   = (float*)    alloc((size_t)NNn * 4);
  float*     adG   = (float*)    alloc((size_t)GGg * 4);
  float*     Ttab  = (float*)    alloc((size_t)15 * DP * 4);
  _Float16*  lin1p = (_Float16*) alloc((size_t)DP * DP * 2);
  _Float16*  W1ap  = (_Float16*) alloc((size_t)DP * DP * 2);
  _Float16*  lin2gp= (_Float16*) alloc((size_t)DP * DP * 2);
  _Float16*  gatWp = (_Float16*) alloc((size_t)4 * DP * DP * 2);
  _Float16*  molWsp= (_Float16*) alloc((size_t)DP * DP * 2);
  _Float16*  molWdp= (_Float16*) alloc((size_t)DP * DP * 2);
  _Float16*  gWihp = (_Float16*) alloc((size_t)5 * D3P * DP * 2);
  _Float16*  gWhhp = (_Float16*) alloc((size_t)5 * D3P * DP * 2);
  _Float16*  mWihp = (_Float16*) alloc((size_t)D3P * DP * 2);
  _Float16*  mWhhp = (_Float16*) alloc((size_t)D3P * DP * 2);
  float*     lin1bp= (float*)    alloc((size_t)DP * 4);
  float*     gebp  = (float*)    alloc((size_t)DP * 4);
  (void)ws_size; (void)n_in; (void)in_sizes; (void)out_size;

  auto gemm = [&](const _Float16* A, const _Float16* W, float* C,
                  int M, int Nout, const float* bias, int act) {
    int total = (M / 16) * (Nout / 64);
    k_gemm_wmma<<<(total + 7) / 8, 256, 0, stream>>>(A, W, C, M, DP, Nout, bias, act);
  };
  auto gru = [&](const _Float16* ein, const _Float16* h16, const float* h32,
                 const _Float16* Wih, const _Float16* Whh,
                 const float* bi, const float* bh,
                 float* o32, _Float16* o16, int M) {
    int total = (M / 16) * (DP / 16);
    k_gru_fused<<<(total + 7) / 8, 256, 0, stream>>>(ein, h16, h32, Wih, Whh,
                                                     bi, bh, o32, o16, M);
  };

  // ---- pack weights to padded f16 ----
  k_pack_w<<<blks((long)DP * DP), 256, 0, stream>>>(lin1p,  lin1_W,    Dv, 0, 1);
  k_pack_w<<<blks((long)DP * DP), 256, 0, stream>>>(W1ap,   ge_lin1_W, 2 * Dv, 0, 1);
  k_pack_w<<<blks((long)DP * DP), 256, 0, stream>>>(lin2gp, ge_lin2_W, Dv, 0, 1);
  for (int l = 0; l < 4; ++l)
    k_pack_w<<<blks((long)DP * DP), 256, 0, stream>>>(gatWp + (size_t)l * DP * DP,
                                                      gat_W + (size_t)l * Dv * Dv, Dv, 0, 1);
  for (int l = 0; l < 5; ++l) {
    k_pack_w<<<blks((long)3 * DP * DP), 256, 0, stream>>>(gWihp + (size_t)l * D3P * DP,
                                                          gru_wih + (size_t)l * 3 * Dv * Dv, Dv, 0, 3);
    k_pack_w<<<blks((long)3 * DP * DP), 256, 0, stream>>>(gWhhp + (size_t)l * D3P * DP,
                                                          gru_whh + (size_t)l * 3 * Dv * Dv, Dv, 0, 3);
  }
  k_pack_w<<<blks((long)DP * DP), 256, 0, stream>>>(molWsp, mol_Ws, Dv, 0, 1);
  k_pack_w<<<blks((long)DP * DP), 256, 0, stream>>>(molWdp, mol_Wd, Dv, 0, 1);
  k_pack_w<<<blks((long)3 * DP * DP), 256, 0, stream>>>(mWihp, mgru_wih, Dv, 0, 3);
  k_pack_w<<<blks((long)3 * DP * DP), 256, 0, stream>>>(mWhhp, mgru_whh, Dv, 0, 3);
  k_pad_vec<<<blks(DP), 256, 0, stream>>>(lin1bp, lin1_b, Dv, DP);
  k_pad_vec<<<blks(DP), 256, 0, stream>>>(gebp,   ge_bias, Dv, DP);

  float*    XF32 = XF32a;  _Float16* XF16 = XF16a;   // current node state
  float*    XF32n = XF32b; _Float16* XF16n = XF16b;  // next

  // ---- embed + lin1 ----
  k_embed<<<blks((long)NNn * DP), 256, 0, stream>>>(x, emb_atom, emb_chir, XF16);
  gemm(XF16, lin1p, XF32, NNn, DP, lin1bp, /*leaky*/1);
  k_to_f16<<<blks((long)NNn * DP), 256, 0, stream>>>(XF32, nullptr, XF16, NNn, 0);

  const int* ei_src = edge_index;
  const int* ei_dst = edge_index + EEe;

  // ---- layer 0: GATEConv ----
  gemm(XF16, W1ap, P32, NNn, DP, nullptr, 0);                         // P = xf @ W1a^T
  k_table_T<<<blks(15 * DP), 256, 0, stream>>>(ge_e1, ge_e2, ge_lin1_W, Ttab);
  k_rowdot<<<blks(NNn), 256, 0, stream>>>(XF32, ge_att_r, asN, NNn);  // ar
  k_fill<<<blks(NNn), 256, 0, stream>>>(mseg, -INFINITY, NNn);
  k_fill<<<blks(NNn), 256, 0, stream>>>(sseg, 0.f, NNn);
  k_edge0_alpha<<<blks(NEc), 256, 0, stream>>>(edge_index, edge_attr, P32, Ttab,
                                               asN, ge_att_l, XJ, aE, dst0, mseg);
  k_edge_exp<<<blks(NEc), 256, 0, stream>>>(aE, dst0, mseg, sseg, eEx, NEc);
  k_fill<<<blks((long)NNn * DP), 256, 0, stream>>>(S32, 0.f, (long)NNn * DP);
  k_accum_direct16<<<blks((long)NEc * DP), 256, 0, stream>>>(dst0, eEx, sseg, XJ, S32, NEc);
  k_to_f16<<<blks((long)NNn * DP), 256, 0, stream>>>(S32, nullptr, TMP16, NNn, 0);
  gemm(TMP16, lin2gp, H0, NNn, DP, gebp, 0);                          // h = S @ L2^T + b
  k_to_f16<<<blks((long)NNn * DP), 256, 0, stream>>>(H0, nullptr, TMP16, NNn, /*elu*/2);
  gru(TMP16, XF16, XF32, gWihp, gWhhp, gru_bih, gru_bhh, XF32n, XF16n, NNn);
  { float* t = XF32; XF32 = XF32n; XF32n = t; }
  { _Float16* t = XF16; XF16 = XF16n; XF16n = t; }

  // ---- layers 1..4: GATConv ----
  for (int l = 0; l < 4; ++l) {
    gemm(XF16, gatWp + (size_t)l * DP * DP, P32, NNn, DP, nullptr, 0);       // xw
    k_rowdot<<<blks(NNn), 256, 0, stream>>>(P32, gat_att_s + l * Dv, asN, NNn);
    k_rowdot<<<blks(NNn), 256, 0, stream>>>(P32, gat_att_d + l * Dv, adN, NNn);
    k_fill<<<blks(NNn), 256, 0, stream>>>(mseg, -INFINITY, NNn);
    k_fill<<<blks(NNn), 256, 0, stream>>>(sseg, 0.f, NNn);
    k_edge_max<<<blks(EEe), 256, 0, stream>>>(ei_src, ei_dst, asN, adN, aE, mseg, EEe, 0.2f);
    k_edge_exp<<<blks(EEe), 256, 0, stream>>>(aE, ei_dst, mseg, sseg, eEx, EEe);
    k_fill<<<blks((long)NNn * DP), 256, 0, stream>>>(S32, 0.f, (long)NNn * DP);
    k_accum_gather<<<blks((long)EEe * DP), 256, 0, stream>>>(ei_src, ei_dst, eEx, sseg,
                                                             P32, S32, EEe);
    k_to_f16<<<blks((long)NNn * DP), 256, 0, stream>>>(S32, gat_b + l * Dv, TMP16, NNn, 2);
    gru(TMP16, XF16, XF32,
        gWihp + (size_t)(l + 1) * D3P * DP, gWhhp + (size_t)(l + 1) * D3P * DP,
        gru_bih + (l + 1) * 3 * Dv, gru_bhh + (l + 1) * 3 * Dv, XF32n, XF16n, NNn);
    { float* t = XF32; XF32 = XF32n; XF32n = t; }
    { _Float16* t = XF16; XF16 = XF16n; XF16n = t; }
  }

  // ---- attentive readout ----
  gemm(XF16, molWsp, P32, NNn, DP, nullptr, 0);                       // xs (fixed)
  k_rowdot<<<blks(NNn), 256, 0, stream>>>(P32, mol_att_s, asN, NNn);  // as_node (fixed)
  float*    O32 = O32a;  _Float16* O16 = O16a;
  float*    O32n = O32b; _Float16* O16n = O16b;
  k_fill<<<blks((long)GGg * DP), 256, 0, stream>>>(O32, 0.f, (long)GGg * DP);
  k_accum_nodes<<<blks((long)NNn * DP), 256, 0, stream>>>(batch, XF32, O32);
  k_relu_pad<<<blks((long)GGg * DP), 256, 0, stream>>>(O32, GGg);

  for (int it = 0; it < 3; ++it) {
    k_to_f16<<<blks((long)GGg * DP), 256, 0, stream>>>(O32, nullptr, O16, GGg, 0);
    gemm(O16, molWdp, XD, GGg, DP, nullptr, 0);                       // xd
    k_rowdot<<<blks(GGg), 256, 0, stream>>>(XD, mol_att_d, adG, GGg);
    k_fill<<<blks(GGg), 256, 0, stream>>>(mseg, -INFINITY, GGg);
    k_fill<<<blks(GGg), 256, 0, stream>>>(sseg, 0.f, GGg);
    k_edge_max<<<blks(NNn), 256, 0, stream>>>(nullptr, batch, asN, adG, aE, mseg, NNn, 0.2f);
    k_edge_exp<<<blks(NNn), 256, 0, stream>>>(aE, batch, mseg, sseg, eEx, NNn);
    k_fill<<<blks((long)GGg * DP), 256, 0, stream>>>(S32, 0.f, (long)GGg * DP);
    k_accum_gather<<<blks((long)NNn * DP), 256, 0, stream>>>(nullptr, batch, eEx, sseg,
                                                             P32, S32, NNn);
    k_to_f16<<<blks((long)GGg * DP), 256, 0, stream>>>(S32, mol_b, HG16, GGg, 2);
    gru(HG16, O16, O32, mWihp, mWhhp, mgru_bih, mgru_bhh, O32n, O16n, GGg);
    { float* t = O32; O32 = O32n; O32n = t; }
    { _Float16* t = O16; O16 = O16n; O16n = t; }
  }

  k_writeout<<<blks((long)GGg * DP), 256, 0, stream>>>(O32, out);
  k_pred<<<blks(GGg), 256, 0, stream>>>(O32, lin2_W, lin2_b, out);
}
